// AttentionPairBias_5789615915154
// MI455X (gfx1250) — compile-verified, matrix operand order checked by reading, not run on hardware
//
#include <hip/hip_runtime.h>
#include <hip/hip_bf16.h>
#include <math.h>

#define DEVI __device__ __forceinline__

typedef __attribute__((ext_vector_type(16))) __bf16 v16bf;
typedef __attribute__((ext_vector_type(8)))  __bf16 v8bf;
typedef __attribute__((ext_vector_type(8)))  float  v8f;

constexpr int B_ = 2, L_ = 1024, CA_ = 512, CS_ = 384, CZ_ = 64, H_ = 16, D_ = 32;
constexpr int ML = B_ * L_;   // 2048 total rows (b*L + l)

DEVI int lane_id() { return threadIdx.x & 31; }
DEVI float sigmoidf(float x) { return 1.f / (1.f + __expf(-x)); }

// ---------------------------------------------------------------------------
// WMMA fragment loaders (CDNA5 ISA 7.12.2, 16-bit A 16x32 / B 32x16 layout)
// A-frag: lane l holds row (l&15); lanes<16: elems0-7=K0..7, elems8-15=K16..23
//          lanes>=16: elems0-7=K8..15, elems8-15=K24..31
// B-frag for x@W.T is the identical pattern applied to rows of W (row=out-ch).
// ---------------------------------------------------------------------------
DEVI v16bf frag_rowmajor(const __bf16* base, int ld, int row0, int k0) {
  int l = lane_id(); int r = l & 15; int hf = l >> 4;
  const __bf16* p = base + (size_t)(row0 + r) * ld + (k0 + hf * 8);
  v8bf lo = *(const v8bf*)(p);
  v8bf hi = *(const v8bf*)(p + 16);
  v16bf f;
#pragma unroll
  for (int i = 0; i < 8; ++i) { f[i] = lo[i]; f[i + 8] = hi[i]; }
  return f;
}

// B-frag where the K dimension is strided in memory (V matrix: B[kk,d]=V[k0+kk,d])
DEVI v16bf frag_kstrided(const __bf16* base, int ld, int k0, int col) {
  int hf = lane_id() >> 4;
  v16bf f;
#pragma unroll
  for (int j = 0; j < 8; ++j) {
    f[j]     = base[(size_t)(k0 + hf * 8 + j) * ld + col];
    f[j + 8] = base[(size_t)(k0 + 16 + hf * 8 + j) * ld + col];
  }
  return f;
}

DEVI v8f wmma_bf16(v16bf a, v16bf b, v8f c) {
  // (neg_a, A, neg_b, B, c_mod, C, reuse_a, reuse_b)
  return __builtin_amdgcn_wmma_f32_16x16x32_bf16(false, a, false, b, (short)0, c,
                                                 false, false);
}

// ---------------------------------------------------------------------------
// K0: elementwise converts / layernorms
// ---------------------------------------------------------------------------
__global__ void f32_to_bf16_kernel(const float* __restrict__ src,
                                   __bf16* __restrict__ dst, int n) {
  for (int i = blockIdx.x * blockDim.x + threadIdx.x; i < n;
       i += gridDim.x * blockDim.x)
    dst[i] = (__bf16)src[i];
}

__global__ void prep_s_kernel(const float* __restrict__ s,
                              const float* __restrict__ scale,
                              __bf16* __restrict__ s_ln,
                              __bf16* __restrict__ s_raw) {
  int row = blockIdx.x;
  const float* x = s + (size_t)row * CS_;
  __shared__ float sh1[128], sh2[128];
  int t = threadIdx.x;
  float lsum = 0.f, lsq = 0.f;
  for (int i = t; i < CS_; i += 128) { float v = x[i]; lsum += v; lsq += v * v; }
  sh1[t] = lsum; sh2[t] = lsq; __syncthreads();
  for (int o = 64; o > 0; o >>= 1) {
    if (t < o) { sh1[t] += sh1[t + o]; sh2[t] += sh2[t + o]; }
    __syncthreads();
  }
  float mean = sh1[0] * (1.f / CS_);
  float rstd = rsqrtf(sh2[0] * (1.f / CS_) - mean * mean + 1e-5f);
  for (int i = t; i < CS_; i += 128) {
    float xv = x[i];
    s_ln[(size_t)row * CS_ + i]  = (__bf16)((xv - mean) * rstd * scale[i]);
    s_raw[(size_t)row * CS_ + i] = (__bf16)xv;
  }
}

__global__ void prep_a_kernel(const float* __restrict__ a,
                              float* __restrict__ a_ln) {
  int row = blockIdx.x;
  const float* x = a + (size_t)row * CA_;
  __shared__ float sh1[128], sh2[128];
  int t = threadIdx.x;
  float lsum = 0.f, lsq = 0.f;
  for (int i = t; i < CA_; i += 128) { float v = x[i]; lsum += v; lsq += v * v; }
  sh1[t] = lsum; sh2[t] = lsq; __syncthreads();
  for (int o = 64; o > 0; o >>= 1) {
    if (t < o) { sh1[t] += sh1[t + o]; sh2[t] += sh2[t + o]; }
    __syncthreads();
  }
  float mean = sh1[0] * (1.f / CA_);
  float rstd = rsqrtf(sh2[0] * (1.f / CA_) - mean * mean + 1e-5f);
  for (int i = t; i < CA_; i += 128)
    a_ln[(size_t)row * CA_ + i] = (x[i] - mean) * rstd;
}

// ---------------------------------------------------------------------------
// K1: AdaLN — two GEMMs sharing the A fragment + sigmoid gate epilogue
// a_norm = sigmoid(s_ln@Ws.T + b) * a_ln + s_ln@Wnb.T      (bf16 out)
// ---------------------------------------------------------------------------
__global__ void __launch_bounds__(128)
ada_gemm_kernel(const __bf16* __restrict__ sln, const __bf16* __restrict__ w_s,
                const __bf16* __restrict__ w_nb, const float* __restrict__ ada_b,
                const float* __restrict__ a_ln, __bf16* __restrict__ anorm) {
  int m0 = blockIdx.x * 16;
  int n0 = (blockIdx.y * 4 + (threadIdx.x >> 5)) * 16;
  v8f c1 = {}, c2 = {};
  for (int k0 = 0; k0 < CS_; k0 += 32) {
    v16bf af = frag_rowmajor(sln, CS_, m0, k0);
    c1 = wmma_bf16(af, frag_rowmajor(w_s, CS_, n0, k0), c1);
    c2 = wmma_bf16(af, frag_rowmajor(w_nb, CS_, n0, k0), c2);
  }
  int l = lane_id(); int n = n0 + (l & 15); int hf = l >> 4;
#pragma unroll
  for (int v = 0; v < 8; ++v) {
    int m = m0 + v + 8 * hf;
    float g = sigmoidf(c1[v] + ada_b[n]);
    anorm[(size_t)m * CA_ + n] = (__bf16)(g * a_ln[(size_t)m * CA_ + n] + c2[v]);
  }
}

// ---------------------------------------------------------------------------
// K2: fused Q/K/V/Gate projections.  N-space = 4*CA, group selects weight.
// q/k/v written in [B,H,L,D] bf16; gate = sigmoid(.) fp32 [B,L,CA]
// ---------------------------------------------------------------------------
__global__ void __launch_bounds__(128)
qkvg_kernel(const __bf16* __restrict__ anorm,
            const __bf16* __restrict__ wq, const __bf16* __restrict__ wk,
            const __bf16* __restrict__ wv, const __bf16* __restrict__ wg,
            const float* __restrict__ bq, const float* __restrict__ bk,
            const float* __restrict__ bv, const float* __restrict__ bg,
            __bf16* __restrict__ q, __bf16* __restrict__ k,
            __bf16* __restrict__ v, float* __restrict__ gate) {
  int m0 = blockIdx.x * 16;
  int ntg = blockIdx.y * 4 + (threadIdx.x >> 5);   // 0..127
  int grp = ntg >> 5;                              // which projection
  int n0 = (ntg & 31) * 16;
  const __bf16* w = (grp == 0) ? wq : (grp == 1) ? wk : (grp == 2) ? wv : wg;
  const float* bias = (grp == 0) ? bq : (grp == 1) ? bk : (grp == 2) ? bv : bg;
  v8f acc = {};
  for (int k0 = 0; k0 < CA_; k0 += 32)
    acc = wmma_bf16(frag_rowmajor(anorm, CA_, m0, k0),
                    frag_rowmajor(w, CA_, n0, k0), acc);
  int l = lane_id(); int n = n0 + (l & 15); int hf = l >> 4;
#pragma unroll
  for (int vv = 0; vv < 8; ++vv) {
    int m = m0 + vv + 8 * hf;
    int bb = m >> 10, ll = m & (L_ - 1);
    float val = acc[vv] + bias[n];
    if (grp == 3) {
      gate[(size_t)m * CA_ + n] = sigmoidf(val);
    } else {
      int h = n >> 5, d = n & 31;
      __bf16* dst = (grp == 0) ? q : (grp == 1) ? k : v;
      dst[(((size_t)bb * H_ + h) * L_ + ll) * D_ + d] = (__bf16)val;
    }
  }
}

// ---------------------------------------------------------------------------
// K3: pair bias.  Per wave: LN 16 rows of z (CZ=64), pack directly into the
// ISA A-fragment layout, x wz.T via 2 WMMAs -> bias[B,H,L,L] (bf16).
// z read exactly once; bf16 output halves bias HBM traffic (128MB -> 64MB).
// ---------------------------------------------------------------------------
__global__ void __launch_bounds__(128)
pair_bias_kernel(const float* __restrict__ z, const float* __restrict__ zscale,
                 const __bf16* __restrict__ wzb, __bf16* __restrict__ biasOut) {
  int b = blockIdx.z, i = blockIdx.y;
  int jt = blockIdx.x * 4 + (threadIdx.x >> 5);
  int l = lane_id(); int r = l & 15; int hf = l >> 4;
  int j = jt * 16 + r;
  const float* zr = z + (((size_t)b * L_ + i) * L_ + j) * CZ_;
  float vals[32]; float sum = 0.f, sq = 0.f;
#pragma unroll
  for (int c = 0; c < 4; ++c) {
    int kb = c * 16 + hf * 8;
#pragma unroll
    for (int jj = 0; jj < 8; ++jj) {
      float xv = zr[kb + jj];
      vals[c * 8 + jj] = xv; sum += xv; sq += xv * xv;
    }
  }
  // paired lanes (r, r+16) hold complementary halves of the same row
  sum += __shfl_xor(sum, 16, 32);
  sq  += __shfl_xor(sq, 16, 32);
  float mean = sum * (1.f / 64.f);
  float rstd = rsqrtf(sq * (1.f / 64.f) - mean * mean + 1e-5f);
  v16bf a0, a1;
#pragma unroll
  for (int c = 0; c < 4; ++c) {
    int kb = c * 16 + hf * 8;
#pragma unroll
    for (int jj = 0; jj < 8; ++jj) {
      float yv = (vals[c * 8 + jj] - mean) * rstd * zscale[kb + jj];
      int slot = (c & 1) * 8 + jj;
      if (c < 2) a0[slot] = (__bf16)yv; else a1[slot] = (__bf16)yv;
    }
  }
  v8f acc = {};
  acc = wmma_bf16(a0, frag_rowmajor(wzb, CZ_, 0, 0),  acc);
  acc = wmma_bf16(a1, frag_rowmajor(wzb, CZ_, 0, 32), acc);
  int hd = l & 15;                         // head column
#pragma unroll
  for (int v = 0; v < 8; ++v) {
    int jj = jt * 16 + v + 8 * hf;
    biasOut[(((size_t)b * H_ + hd) * L_ + i) * L_ + jj] = (__bf16)acc[v];
  }
}

// ---------------------------------------------------------------------------
// K4: flash attention per (b,h,16-query tile).  32 keys per step:
//   S = Q@K.T (2 WMMAs, K=D=32) + bias, online softmax w/ key mask,
//   P repacked via per-wave LDS into A-frag, out += P@V (2 WMMAs).
// Next K/V tile prefetched (global_prefetch_b8) while this tile computes.
// Epilogue: renormalize, apply gate, write bf16 [B,L,CA].
// ---------------------------------------------------------------------------
__global__ void __launch_bounds__(128)
attn_kernel(const __bf16* __restrict__ qm, const __bf16* __restrict__ km,
            const __bf16* __restrict__ vm, const __bf16* __restrict__ biasT,
            const int* __restrict__ mask, const float* __restrict__ gate,
            __bf16* __restrict__ outg) {
  __shared__ __align__(16) __bf16 plds[4][16][32];
  int b = blockIdx.z, h = blockIdx.y;
  int w = threadIdx.x >> 5;
  int q0 = (blockIdx.x * 4 + w) * 16;
  int l = lane_id(); int n = l & 15; int hf = l >> 4;
  const __bf16* qb = qm + ((size_t)b * H_ + h) * L_ * D_;
  const __bf16* kb = km + ((size_t)b * H_ + h) * L_ * D_;
  const __bf16* vb = vm + ((size_t)b * H_ + h) * L_ * D_;
  const __bf16* bptr = biasT + (((size_t)b * H_ + h) * L_ + q0) * L_;
  const int* mk = mask + b * L_;
  v16bf qfrag = frag_rowmajor(qb, D_, q0, 0);
  float mrun[8], lrun[8];
#pragma unroll
  for (int v = 0; v < 8; ++v) { mrun[v] = -1e30f; lrun[v] = 0.f; }
  v8f o0 = {}, o1 = {};
  const float scale = 0.17677669529663687f;   // 1/sqrt(32)

  for (int k0 = 0; k0 < L_; k0 += 32) {
    // prefetch next 32-key tile of K and V (each lane covers 64B -> 2KB/tile)
    if (k0 + 32 < L_) {
      __builtin_prefetch(kb + (size_t)(k0 + 32) * D_ + l * 32, 0, 1);
      __builtin_prefetch(vb + (size_t)(k0 + 32) * D_ + l * 32, 0, 1);
    }
    v8f zero = {};
    v8f s0 = wmma_bf16(qfrag, frag_rowmajor(kb, D_, k0, 0), zero);
    v8f s1 = wmma_bf16(qfrag, frag_rowmajor(kb, D_, k0 + 16, 0), zero);
    int kc0 = k0 + n, kc1 = k0 + 16 + n;
    bool m0 = mk[kc0] != 0, m1 = mk[kc1] != 0;
    float e0[8], e1[8], tmax[8], lsum[8];
#pragma unroll
    for (int v = 0; v < 8; ++v) {
      int mrow = v + 8 * hf;
      e0[v] = s0[v] * scale + (float)bptr[(size_t)mrow * L_ + kc0];
      e1[v] = s1[v] * scale + (float)bptr[(size_t)mrow * L_ + kc1];
      tmax[v] = fmaxf(m0 ? e0[v] : -1e30f, m1 ? e1[v] : -1e30f);
    }
#pragma unroll
    for (int off = 1; off < 16; off <<= 1)
#pragma unroll
      for (int v = 0; v < 8; ++v)
        tmax[v] = fmaxf(tmax[v], __shfl_xor(tmax[v], off, 32));
#pragma unroll
    for (int v = 0; v < 8; ++v) {
      float mnew = fmaxf(mrun[v], tmax[v]);
      float alpha = __expf(mrun[v] - mnew);
      mrun[v] = mnew;
      float p0 = m0 ? __expf(e0[v] - mnew) : 0.f;
      float p1 = m1 ? __expf(e1[v] - mnew) : 0.f;
      e0[v] = p0; e1[v] = p1; lsum[v] = p0 + p1;
      o0[v] *= alpha; o1[v] *= alpha; lrun[v] *= alpha;
    }
#pragma unroll
    for (int off = 1; off < 16; off <<= 1)
#pragma unroll
      for (int v = 0; v < 8; ++v) lsum[v] += __shfl_xor(lsum[v], off, 32);
#pragma unroll
    for (int v = 0; v < 8; ++v) lrun[v] += lsum[v];
    // repack P (C-layout) -> A-frag layout through LDS
#pragma unroll
    for (int v = 0; v < 8; ++v) {
      int mrow = v + 8 * hf;
      plds[w][mrow][n]      = (__bf16)e0[v];
      plds[w][mrow][16 + n] = (__bf16)e1[v];
    }
    __syncthreads();
    v16bf pf = frag_rowmajor(&plds[w][0][0], 32, 0, 0);
    o0 = wmma_bf16(pf, frag_kstrided(vb, D_, k0, n), o0);
    o1 = wmma_bf16(pf, frag_kstrided(vb, D_, k0, 16 + n), o1);
    __syncthreads();
  }
#pragma unroll
  for (int v = 0; v < 8; ++v) {
    int m = q0 + v + 8 * hf;
    float inv = 1.f / fmaxf(lrun[v], 1e-6f);
    size_t row = ((size_t)b * L_ + m) * CA_ + (size_t)h * D_;
    outg[row + n]      = (__bf16)(o0[v] * inv * gate[row + n]);
    outg[row + 16 + n] = (__bf16)(o1[v] * inv * gate[row + 16 + n]);
  }
}

// ---------------------------------------------------------------------------
// K5: glast = sigmoid(s @ w_last.T + b_last)   (fp32)
// ---------------------------------------------------------------------------
__global__ void __launch_bounds__(128)
last_gate_kernel(const __bf16* __restrict__ sraw, const __bf16* __restrict__ wl,
                 const float* __restrict__ bl, float* __restrict__ glast) {
  int m0 = blockIdx.x * 16;
  int n0 = (blockIdx.y * 4 + (threadIdx.x >> 5)) * 16;
  v8f acc = {};
  for (int k0 = 0; k0 < CS_; k0 += 32)
    acc = wmma_bf16(frag_rowmajor(sraw, CS_, m0, k0),
                    frag_rowmajor(wl, CS_, n0, k0), acc);
  int l = lane_id(); int n = n0 + (l & 15); int hf = l >> 4;
#pragma unroll
  for (int v = 0; v < 8; ++v) {
    int m = m0 + v + 8 * hf;
    glast[(size_t)m * CA_ + n] = sigmoidf(acc[v] + bl[n]);
  }
}

// ---------------------------------------------------------------------------
// K6: out = glast * (attn_gated @ wo.T + bo) * mask_q
// ---------------------------------------------------------------------------
__global__ void __launch_bounds__(128)
out_proj_kernel(const __bf16* __restrict__ x, const __bf16* __restrict__ wob,
                const float* __restrict__ bo, const float* __restrict__ glast,
                const int* __restrict__ mask, float* __restrict__ out) {
  int m0 = blockIdx.x * 16;
  int n0 = (blockIdx.y * 4 + (threadIdx.x >> 5)) * 16;
  v8f acc = {};
  for (int k0 = 0; k0 < CA_; k0 += 32)
    acc = wmma_bf16(frag_rowmajor(x, CA_, m0, k0),
                    frag_rowmajor(wob, CA_, n0, k0), acc);
  int l = lane_id(); int n = n0 + (l & 15); int hf = l >> 4;
#pragma unroll
  for (int v = 0; v < 8; ++v) {
    int m = m0 + v + 8 * hf;
    float mf = (mask[m] != 0) ? 1.f : 0.f;
    out[(size_t)m * CA_ + n] =
        glast[(size_t)m * CA_ + n] * (acc[v] + bo[n]) * mf;
  }
}

// ---------------------------------------------------------------------------
extern "C" void kernel_launch(void* const* d_in, const int* in_sizes, int n_in,
                              void* d_out, int out_size, void* d_ws, size_t ws_size,
                              hipStream_t stream) {
  const float* a          = (const float*)d_in[0];
  const float* s          = (const float*)d_in[1];
  const float* z          = (const float*)d_in[2];
  const float* ln_s_scale = (const float*)d_in[3];
  const float* ada_w_s    = (const float*)d_in[4];
  const float* ada_b_s    = (const float*)d_in[5];
  const float* ada_w_nb   = (const float*)d_in[6];
  const float* wq = (const float*)d_in[7];  const float* bq = (const float*)d_in[8];
  const float* wk = (const float*)d_in[9];  const float* bk = (const float*)d_in[10];
  const float* wv = (const float*)d_in[11]; const float* bv = (const float*)d_in[12];
  const float* wg = (const float*)d_in[13]; const float* bg = (const float*)d_in[14];
  const float* wo = (const float*)d_in[15]; const float* bo = (const float*)d_in[16];
  const float* ln_z_scale = (const float*)d_in[17];
  const float* wz         = (const float*)d_in[18];
  const float* w_last     = (const float*)d_in[19];
  const float* b_last     = (const float*)d_in[20];
  const int*   mask       = (const int*)d_in[21];
  float* out = (float*)d_out;

  char* base = (char*)d_ws; size_t off = 0;
  auto alloc = [&](size_t bytes) -> void* {
    void* p = base + off;
    off += (bytes + 255) & ~(size_t)255;
    return p;
  };
  __bf16* s_ln_bf  = (__bf16*)alloc((size_t)ML * CS_ * 2);
  __bf16* s_raw_bf = (__bf16*)alloc((size_t)ML * CS_ * 2);
  float*  a_ln_f   = (float*) alloc((size_t)ML * CA_ * 4);
  __bf16* anorm_bf = (__bf16*)alloc((size_t)ML * CA_ * 2);
  __bf16* w_s_bf   = (__bf16*)alloc((size_t)CA_ * CS_ * 2);
  __bf16* w_nb_bf  = (__bf16*)alloc((size_t)CA_ * CS_ * 2);
  __bf16* wq_bf    = (__bf16*)alloc((size_t)CA_ * CA_ * 2);
  __bf16* wk_bf    = (__bf16*)alloc((size_t)CA_ * CA_ * 2);
  __bf16* wv_bf    = (__bf16*)alloc((size_t)CA_ * CA_ * 2);
  __bf16* wg_bf    = (__bf16*)alloc((size_t)CA_ * CA_ * 2);
  __bf16* wo_bf    = (__bf16*)alloc((size_t)CA_ * CA_ * 2);
  __bf16* wl_bf    = (__bf16*)alloc((size_t)CA_ * CS_ * 2);
  __bf16* wz_bf    = (__bf16*)alloc((size_t)H_ * CZ_ * 2);
  __bf16* q_bf     = (__bf16*)alloc((size_t)ML * CA_ * 2);
  __bf16* k_bf     = (__bf16*)alloc((size_t)ML * CA_ * 2);
  __bf16* v_bf     = (__bf16*)alloc((size_t)ML * CA_ * 2);
  float*  gate_f   = (float*) alloc((size_t)ML * CA_ * 4);
  float*  glast_f  = (float*) alloc((size_t)ML * CA_ * 4);
  __bf16* attng_bf = (__bf16*)alloc((size_t)ML * CA_ * 2);
  __bf16* bias_bf  = (__bf16*)alloc((size_t)B_ * H_ * L_ * L_ * 2);   // 64 MB

  auto cvt = [&](const float* src, __bf16* dst, int n) {
    f32_to_bf16_kernel<<<(n + 255) / 256, 256, 0, stream>>>(src, dst, n);
  };
  cvt(ada_w_s,  w_s_bf,  CA_ * CS_);
  cvt(ada_w_nb, w_nb_bf, CA_ * CS_);
  cvt(wq, wq_bf, CA_ * CA_);
  cvt(wk, wk_bf, CA_ * CA_);
  cvt(wv, wv_bf, CA_ * CA_);
  cvt(wg, wg_bf, CA_ * CA_);
  cvt(wo, wo_bf, CA_ * CA_);
  cvt(w_last, wl_bf, CA_ * CS_);
  cvt(wz, wz_bf, H_ * CZ_);

  prep_s_kernel<<<ML, 128, 0, stream>>>(s, ln_s_scale, s_ln_bf, s_raw_bf);
  prep_a_kernel<<<ML, 128, 0, stream>>>(a, a_ln_f);

  ada_gemm_kernel<<<dim3(ML / 16, CA_ / 64), 128, 0, stream>>>(
      s_ln_bf, w_s_bf, w_nb_bf, ada_b_s, a_ln_f, anorm_bf);

  qkvg_kernel<<<dim3(ML / 16, 4 * CA_ / 64), 128, 0, stream>>>(
      anorm_bf, wq_bf, wk_bf, wv_bf, wg_bf, bq, bk, bv, bg,
      q_bf, k_bf, v_bf, gate_f);

  pair_bias_kernel<<<dim3(L_ / 64, L_, B_), 128, 0, stream>>>(
      z, ln_z_scale, wz_bf, bias_bf);

  attn_kernel<<<dim3(L_ / 64, H_, B_), 128, 0, stream>>>(
      q_bf, k_bf, v_bf, bias_bf, mask, gate_f, attng_bf);

  last_gate_kernel<<<dim3(ML / 16, CA_ / 64), 128, 0, stream>>>(
      s_raw_bf, wl_bf, b_last, glast_f);

  out_proj_kernel<<<dim3(ML / 16, CA_ / 64), 128, 0, stream>>>(
      attng_bf, wo_bf, bo, glast_f, mask, out);
}
